// LocalRetro_86440511800124
// MI455X (gfx1250) — compile-verified, matrix operand order checked by reading, not run on hardware
//
#include <hip/hip_runtime.h>
#include <math.h>

// MPNN-GNN (DGL MPNNGNN-style) for MI455X / gfx1250.
// Algorithmic restructure: never materialize the 819MB per-edge weight tensor.
// Per step compute P[n,c,k] = sum_h nf[n,h]*W_e2[c,h*64+k] (82MB f16, L2-resident)
// with WMMA, then a per-edge c-contraction gathers P[src] out of L2.
// GEMM path: K padded to %32, B operands pre-swizzled into WMMA fragment layout,
// 4 independent 16x16 accumulators per wave (no WMMA->WMMA RAW hazards).

#define NN      10000
#define NE      50000
#define NODE_IN 74
#define NODE_KP 96     // padded K for projection GEMM
#define EDGE_IN 12
#define EDGE_KP 32     // padded K for edge-net GEMM
#define HDIM    64
#define NG      128
#define STEPS   6

typedef __attribute__((ext_vector_type(16))) _Float16 v16h;
typedef __attribute__((ext_vector_type(8)))  _Float16 h8;
typedef __attribute__((ext_vector_type(8)))  float    v8f;
typedef __attribute__((ext_vector_type(2)))  _Float16 h2;

union AFrag { v16h v; h8 h[2]; };

// ------------------------- conversion / layout kernels -------------------------

__global__ void mpnn_cvt_f16(const float* __restrict__ s, _Float16* __restrict__ d, int n) {
  int i = blockIdx.x * blockDim.x + threadIdx.x;
  if (i < n) d[i] = (_Float16)s[i];
}

// f32 (rows x cin) -> f16 (rows x cout) with zero pad columns (A operands).
__global__ void mpnn_cvt_pad(const float* __restrict__ s, _Float16* __restrict__ d,
                             int rows, int cin, int cout) {
  int i = blockIdx.x * blockDim.x + threadIdx.x;
  if (i >= rows * cout) return;
  int r = i / cout, c = i % cout;
  d[i] = (c < cin) ? (_Float16)s[(size_t)r * cin + c] : (_Float16)0;
}

// W_e2 is (64 c, 4096 (h,k)); we need B[h, c*64+k] = W_e2[c, h*64+k] for the P GEMM.
__global__ void mpnn_permute_we2(const float* __restrict__ s, _Float16* __restrict__ d) {
  int i = blockIdx.x * blockDim.x + threadIdx.x;   // over 64*4096
  if (i >= 64 * 4096) return;
  int c = i >> 12; int hk = i & 4095; int h = hk >> 6; int k = hk & 63;
  d[h * 4096 + c * 64 + k] = (_Float16)s[i];
}

// W (rows x cols) f32 -> WT (cols x rows) f16  (for GRU weights: x @ W^T)
__global__ void mpnn_cvt_tr(const float* __restrict__ s, _Float16* __restrict__ d,
                            int rows, int cols) {
  int i = blockIdx.x * blockDim.x + threadIdx.x;
  if (i >= rows * cols) return;
  int r = i / cols, c = i % cols;
  d[c * rows + r] = (_Float16)s[i];
}

// Pack row-major f16 B (Korig x N, ld N) into WMMA fragment order:
// Bf[((kb/32)*(N/16) + tn)*512 + lane*16 + e] = B[k, tn*16 + (lane&15)]
// with e = 2v+p, k = kb + (v/4)*16 + (lane/16)*8 + (v%4)*2 + p.  Pad rows -> 0.
__global__ void mpnn_pack_bfrag(const _Float16* __restrict__ B, _Float16* __restrict__ Bf,
                                int Korig, int Kpad, int N) {
  int i = blockIdx.x * blockDim.x + threadIdx.x;
  int total = (Kpad >> 5) * (N >> 4) * 512;
  if (i >= total) return;
  int e = i & 15, lane = (i >> 4) & 31, tile = i >> 9;
  int tilesN = N >> 4;
  int tn = tile % tilesN;
  int kb = (tile / tilesN) << 5;
  int half = lane >> 4, r = lane & 15;
  int v = e >> 1, p = e & 1;
  int k = kb + ((v >> 2) << 4) + (half << 3) + ((v & 3) << 1) + p;
  int n = (tn << 4) + r;
  Bf[i] = (k < Korig) ? B[(size_t)k * N + n] : (_Float16)0;
}

__global__ void mpnn_zero_f32(float* __restrict__ p, int n) {
  int i = blockIdx.x * blockDim.x + threadIdx.x;
  if (i < n) p[i] = 0.0f;
}

__global__ void mpnn_relu_bias_h16(const float* __restrict__ agg, const float* __restrict__ bias,
                                   _Float16* __restrict__ x16, int n) {
  int i = blockIdx.x * blockDim.x + threadIdx.x;
  if (i >= n) return;
  x16[i] = (_Float16)fmaxf(agg[i] + bias[i & (HDIM - 1)], 0.0f);
}

// ------------------------- WMMA GEMM: C = act(A @ Bfrag + bias) -------------------------
// A: (M, K) f16 row-major (lda = K, K%32==0, zero-padded). Bf: fragment-packed (K x N).
// One wave computes a 16x64 strip: 4 independent 16x16 accumulators.
// M%16==0, N%64==0. All fragment loads unconditional; EXEC all-ones at WMMA.
__global__ __launch_bounds__(256)
void mpnn_gemm_wmma(const _Float16* __restrict__ A, int lda,
                    const _Float16* __restrict__ Bf,
                    float* __restrict__ Cf, _Float16* __restrict__ Ch,
                    int M, int N, int K,
                    const float* __restrict__ bias, int relu) {
  const int groupsN = N >> 6;
  const int groups  = (M >> 4) * groupsN;
  const int g = blockIdx.x * (blockDim.x >> 5) + (threadIdx.x >> 5);
  if (g >= groups) return;              // wave-uniform exit
  const int gm   = g / groupsN;
  const int gn   = g % groupsN;
  const int lane = threadIdx.x & 31;
  const int r    = lane & 15;
  const int half = lane >> 4;

  const _Float16* __restrict__ Ap = A + (size_t)((gm << 4) + r) * lda + (half << 3);
  const int tilesN = N >> 4;
  const _Float16* __restrict__ Bp = Bf + ((size_t)(gn << 2)) * 512 + lane * 16;
  const size_t bstride = (size_t)tilesN * 512;   // fragment stride per 32-k block

  v8f acc0 = {}, acc1 = {}, acc2 = {}, acc3 = {};
  for (int kb = 0; kb < K; kb += 32) {
    AFrag a;
    a.h[0] = *(const h8*)(Ap);          // halves kb+half*8   .. +7
    a.h[1] = *(const h8*)(Ap + 16);     // halves kb+16+half*8 .. +7
    Ap += 32;
    const v16h b0 = *(const v16h*)(Bp);
    const v16h b1 = *(const v16h*)(Bp + 512);
    const v16h b2 = *(const v16h*)(Bp + 1024);
    const v16h b3 = *(const v16h*)(Bp + 1536);
    Bp += bstride;
    acc0 = __builtin_amdgcn_wmma_f32_16x16x32_f16(false, a.v, false, b0, (short)0, acc0, false, false);
    acc1 = __builtin_amdgcn_wmma_f32_16x16x32_f16(false, a.v, false, b1, (short)0, acc1, false, false);
    acc2 = __builtin_amdgcn_wmma_f32_16x16x32_f16(false, a.v, false, b2, (short)0, acc2, false, false);
    acc3 = __builtin_amdgcn_wmma_f32_16x16x32_f16(false, a.v, false, b3, (short)0, acc3, false, false);
  }

  const int m0 = (gm << 4) + (half << 3);
  const int n0 = (gn << 6) + r;
  v8f accs[4] = {acc0, acc1, acc2, acc3};
#pragma unroll
  for (int j = 0; j < 4; ++j) {
    const int n = n0 + (j << 4);
    const float bb = bias ? bias[n] : 0.0f;
#pragma unroll
    for (int i = 0; i < 8; ++i) {       // C/D layout: VGPR i -> M = half*8 + i
      float val = accs[j][i] + bb;
      if (relu) val = fmaxf(val, 0.0f);
      const size_t idx = (size_t)(m0 + i) * N + n;
      if (Cf) Cf[idx] = val;
      if (Ch) Ch[idx] = (_Float16)val;
    }
  }
}

// --------------- per-edge contraction + scatter: agg[dst] += ehid[e,:] . P[src] ---------------
// One wave per edge; lane owns output cols 2*lane, 2*lane+1. Each c-iteration reads
// one full 128B row of P[src] (perfectly coalesced, L2-resident).
__global__ __launch_bounds__(256)
void mpnn_edge_msg(const _Float16* __restrict__ eh, const _Float16* __restrict__ P,
                   const float* __restrict__ Q, const int* __restrict__ src,
                   const int* __restrict__ dst, float* __restrict__ agg, int E) {
  const int e = blockIdx.x * (blockDim.x >> 5) + (threadIdx.x >> 5);
  if (e >= E) return;
  const int lane = threadIdx.x & 31;
  const int col  = lane << 1;
  const int s = src[e];
  const int d = dst[e];
  const _Float16* __restrict__ Pp = P  + (size_t)s * (HDIM * HDIM) + col;
  const _Float16* __restrict__ ev = eh + (size_t)e * HDIM;

  float acc0 = Q[(size_t)s * HDIM + col];
  float acc1 = Q[(size_t)s * HDIM + col + 1];
#pragma unroll 8
  for (int c = 0; c < HDIM; ++c) {
    const float w = (float)ev[c];
    const h2 p = *(const h2*)(Pp + c * HDIM);
    acc0 += w * (float)p[0];
    acc1 += w * (float)p[1];
  }
  atomicAdd(&agg[(size_t)d * HDIM + col],     acc0);
  atomicAdd(&agg[(size_t)d * HDIM + col + 1], acc1);
}

// ------------------------- GRU gates (torch order r,z,n) -------------------------
__global__ void mpnn_gru_gate(const float* __restrict__ gi, const float* __restrict__ gh,
                              float* __restrict__ h32, _Float16* __restrict__ h16, int n) {
  int i = blockIdx.x * blockDim.x + threadIdx.x;
  if (i >= n) return;
  const int node = i >> 6, k = i & 63;
  const size_t base = (size_t)node * 192;
  const float ir = gi[base + k],        hr = gh[base + k];
  const float iz = gi[base + 64 + k],   hz = gh[base + 64 + k];
  const float in_ = gi[base + 128 + k], hn = gh[base + 128 + k];
  const float rg = 1.0f / (1.0f + __expf(-(ir + hr)));
  const float zg = 1.0f / (1.0f + __expf(-(iz + hz)));
  const float ng = tanhf(in_ + rg * hn);
  const float h  = h32[i];
  const float o  = (1.0f - zg) * ng + zg * h;
  h32[i] = o;
  h16[i] = (_Float16)o;
}

__global__ void mpnn_pool(const float* __restrict__ h32, const int* __restrict__ gid,
                          float* __restrict__ out, int n) {
  int i = blockIdx.x * blockDim.x + threadIdx.x;
  if (i >= n) return;
  const int node = i >> 6, k = i & 63;
  atomicAdd(&out[(size_t)gid[node] * HDIM + k], h32[i]);
}

// ------------------------- host orchestration -------------------------

extern "C" void kernel_launch(void* const* d_in, const int* in_sizes, int n_in,
                              void* d_out, int out_size, void* d_ws, size_t ws_size,
                              hipStream_t stream) {
  const float* node_feats = (const float*)d_in[0];
  const float* edge_feats = (const float*)d_in[1];
  const int*   src        = (const int*)d_in[2];
  const int*   dst        = (const int*)d_in[3];
  const int*   gid        = (const int*)d_in[4];
  const float* W_proj     = (const float*)d_in[5];
  const float* b_proj     = (const float*)d_in[6];
  const float* W_e1       = (const float*)d_in[7];
  const float* b_e1       = (const float*)d_in[8];
  const float* W_e2       = (const float*)d_in[9];
  const float* b_e2       = (const float*)d_in[10];
  const float* conv_bias  = (const float*)d_in[11];
  const float* W_ih       = (const float*)d_in[12];
  const float* W_hh       = (const float*)d_in[13];
  const float* b_ih       = (const float*)d_in[14];
  const float* b_hh       = (const float*)d_in[15];

  // ---- workspace carve-out (~120 MB) ----
  char* w = (char*)d_ws;
  auto take = [&](size_t bytes) -> char* {
    char* p = w; w += (bytes + 255) & ~(size_t)255; return p;
  };
  float*    h32   = (float*)   take((size_t)NN * HDIM * 4);         // hidden (== nf)
  _Float16* h16   = (_Float16*)take((size_t)NN * HDIM * 2);         // f16 mirror
  _Float16* x16   = (_Float16*)take((size_t)NN * HDIM * 2);         // relu(agg+bias)
  float*    agg   = (float*)   take((size_t)NN * HDIM * 4);
  float*    gi    = (float*)   take((size_t)NN * 3 * HDIM * 4);
  float*    gh    = (float*)   take((size_t)NN * 3 * HDIM * 4);
  float*    Qb    = (float*)   take((size_t)NN * HDIM * 4);         // nf @ reshape(b_e2)
  _Float16* eh16  = (_Float16*)take((size_t)NE * HDIM * 2);
  _Float16* P16   = (_Float16*)take((size_t)NN * HDIM * HDIM * 2);  // 82 MB, L2-resident
  _Float16* nf16i = (_Float16*)take((size_t)NN * NODE_KP * 2);      // padded A
  _Float16* ef16  = (_Float16*)take((size_t)NE * EDGE_KP * 2);      // padded A
  // row-major f16 staging for B operands
  _Float16* Wp16  = (_Float16*)take((size_t)NODE_IN * HDIM * 2);
  _Float16* We116 = (_Float16*)take((size_t)EDGE_IN * HDIM * 2);
  _Float16* We216 = (_Float16*)take((size_t)HDIM * HDIM * HDIM * 2);
  _Float16* B216  = (_Float16*)take((size_t)HDIM * HDIM * 2);
  _Float16* WihT  = (_Float16*)take((size_t)HDIM * 3 * HDIM * 2);
  _Float16* WhhT  = (_Float16*)take((size_t)HDIM * 3 * HDIM * 2);
  // fragment-packed B operands
  _Float16* WpF   = (_Float16*)take((size_t)NODE_KP * HDIM * 2);
  _Float16* We1F  = (_Float16*)take((size_t)EDGE_KP * HDIM * 2);
  _Float16* We2F  = (_Float16*)take((size_t)HDIM * HDIM * HDIM * 2);
  _Float16* B2F   = (_Float16*)take((size_t)HDIM * HDIM * 2);
  _Float16* WihF  = (_Float16*)take((size_t)HDIM * 3 * HDIM * 2);
  _Float16* WhhF  = (_Float16*)take((size_t)HDIM * 3 * HDIM * 2);
  (void)ws_size; (void)in_sizes; (void)n_in; (void)out_size;

  auto blk = [](int n) { return (n + 255) / 256; };
  auto gemm = [&](const _Float16* A, int lda, const _Float16* Bf,
                  float* Cf, _Float16* Ch, int M, int N, int K,
                  const float* bias, int relu) {
    const int groups = (M >> 4) * (N >> 6);
    const int blocks = (groups + 7) >> 3;       // 8 waves per 256-thread block
    mpnn_gemm_wmma<<<blocks, 256, 0, stream>>>(A, lda, Bf, Cf, Ch, M, N, K, bias, relu);
  };
  auto pack = [&](const _Float16* B, _Float16* Bf, int Korig, int Kpad, int N) {
    mpnn_pack_bfrag<<<blk(Kpad * N), 256, 0, stream>>>(B, Bf, Korig, Kpad, N);
  };

  // ---- per-call conversions / layouts (cheap; weights are launch-constant) ----
  mpnn_cvt_pad<<<blk(NN * NODE_KP), 256, 0, stream>>>(node_feats, nf16i, NN, NODE_IN, NODE_KP);
  mpnn_cvt_pad<<<blk(NE * EDGE_KP), 256, 0, stream>>>(edge_feats, ef16, NE, EDGE_IN, EDGE_KP);
  mpnn_cvt_f16<<<blk(NODE_IN * HDIM), 256, 0, stream>>>(W_proj, Wp16, NODE_IN * HDIM);
  mpnn_cvt_f16<<<blk(EDGE_IN * HDIM), 256, 0, stream>>>(W_e1, We116, EDGE_IN * HDIM);
  mpnn_permute_we2<<<blk(64 * 4096), 256, 0, stream>>>(W_e2, We216);
  mpnn_cvt_f16<<<blk(4096), 256, 0, stream>>>(b_e2, B216, 4096);       // (h,k) reshape
  mpnn_cvt_tr<<<blk(192 * 64), 256, 0, stream>>>(W_ih, WihT, 192, 64);
  mpnn_cvt_tr<<<blk(192 * 64), 256, 0, stream>>>(W_hh, WhhT, 192, 64);

  pack(Wp16,  WpF,  NODE_IN, NODE_KP, HDIM);
  pack(We116, We1F, EDGE_IN, EDGE_KP, HDIM);
  pack(We216, We2F, HDIM, HDIM, 4096);
  pack(B216,  B2F,  HDIM, HDIM, HDIM);
  pack(WihT,  WihF, HDIM, HDIM, 192);
  pack(WhhT,  WhhF, HDIM, HDIM, 192);

  // nf = hidden = relu(node_feats @ W_proj + b_proj)
  gemm(nf16i, NODE_KP, WpF, h32, h16, NN, HDIM, NODE_KP, b_proj, 1);
  // e_hid = relu(edge_feats @ W_e1 + b_e1)
  gemm(ef16, EDGE_KP, We1F, nullptr, eh16, NE, HDIM, EDGE_KP, b_e1, 1);

  for (int step = 0; step < STEPS; ++step) {
    // P[n, c*64+k] = sum_h nf[n,h] * We2perm[h, c*64+k]   (10000x64 @ 64x4096)
    gemm(h16, HDIM, We2F, nullptr, P16, NN, 4096, HDIM, nullptr, 0);
    // Q[n,k] = sum_h nf[n,h] * b_e2[h*64+k]
    gemm(h16, HDIM, B2F, Qb, nullptr, NN, HDIM, HDIM, nullptr, 0);

    mpnn_zero_f32<<<blk(NN * HDIM), 256, 0, stream>>>(agg, NN * HDIM);
    mpnn_edge_msg<<<(NE + 7) / 8, 256, 0, stream>>>(eh16, P16, Qb, src, dst, agg, NE);

    // x = relu(agg + conv_bias) -> f16 (only feeds the gi GEMM)
    mpnn_relu_bias_h16<<<blk(NN * HDIM), 256, 0, stream>>>(agg, conv_bias, x16, NN * HDIM);

    // gi = x @ W_ih^T + b_ih ; gh = h @ W_hh^T + b_hh   (10000x64 @ 64x192)
    gemm(x16, HDIM, WihF, gi, nullptr, NN, 192, HDIM, b_ih, 0);
    gemm(h16, HDIM, WhhF, gh, nullptr, NN, 192, HDIM, b_hh, 0);

    // hidden = GRU(x, hidden); nf = hidden  (in place, same-index safe)
    mpnn_gru_gate<<<blk(NN * HDIM), 256, 0, stream>>>(gi, gh, h32, h16, NN * HDIM);
  }

  // sum-pool per graph
  mpnn_zero_f32<<<blk(NG * HDIM), 256, 0, stream>>>((float*)d_out, NG * HDIM);
  mpnn_pool<<<blk(NN * HDIM), 256, 0, stream>>>(h32, gid, (float*)d_out, NN * HDIM);
}
